// LinearShift_49417893708312
// MI455X (gfx1250) — compile-verified, matrix operand order checked by reading, not run on hardware
//
#include <hip/hip_runtime.h>

typedef __attribute__((ext_vector_type(16))) _Float16 v16h;
typedef __attribute__((ext_vector_type(8)))  _Float16 v8h;
typedef __attribute__((ext_vector_type(4)))  _Float16 v4h;
typedef __attribute__((ext_vector_type(8)))  float    v8f;
typedef int v4i __attribute__((vector_size(16)));   // b128 payload type for async-LDS builtin

#define BM 128
#define BN 128
#define BK 32
#define LDT 40   // padded LDS row stride (elements): 80B rows -> conflict-free b128 frag reads

#define AS1 __attribute__((address_space(1)))
#define AS3 __attribute__((address_space(3)))

#if __has_builtin(__builtin_amdgcn_global_load_async_to_lds_b128)
#define USE_ASYNC_LDS 1
#endif

#if __has_builtin(__builtin_amdgcn_s_wait_asynccnt)
#define ASYNC_WAIT(n) __builtin_amdgcn_s_wait_asynccnt(n)
#else
#define ASYNC_WAIT(n) asm volatile("s_wait_asynccnt %0" ::"i"(n))
#endif

// ---------------- pre-pass: x fp32 -> fp16 ----------------
__global__ __launch_bounds__(256)
void cvt_x_f16(const float* __restrict__ x, _Float16* __restrict__ xh, long n)
{
    long i = ((long)blockIdx.x * 256 + threadIdx.x) * 8;
    if (i + 8 > n) return;
    float4 f0 = *(const float4*)(x + i);
    float4 f1 = *(const float4*)(x + i + 4);
    v8h o;
    o[0] = (_Float16)f0.x; o[1] = (_Float16)f0.y;
    o[2] = (_Float16)f0.z; o[3] = (_Float16)f0.w;
    o[4] = (_Float16)f1.x; o[5] = (_Float16)f1.y;
    o[6] = (_Float16)f1.z; o[7] = (_Float16)f1.w;
    *(v8h*)(xh + i) = o;
}

// ---------------- pre-pass: v = exp2(round(clip(shift))) * sign(round(sign)) ----------------
__device__ __forceinline__ float pow2sgn(float sh, float sg)
{
    float c = fminf(fmaxf(sh, -14.0f), 0.0f);
    float e = __builtin_exp2f(rintf(c));          // exact power of two
    float r = rintf(sg);
    float s = (r > 0.0f) ? 1.0f : ((r < 0.0f) ? -1.0f : 0.0f);
    return e * s;
}

__global__ __launch_bounds__(256)
void make_v_f16(const float* __restrict__ shift, const float* __restrict__ sgn,
                _Float16* __restrict__ vh, long n)
{
    long i = ((long)blockIdx.x * 256 + threadIdx.x) * 4;
    if (i + 4 > n) return;
    float4 sh = *(const float4*)(shift + i);
    float4 sg = *(const float4*)(sgn + i);
    v4h o;
    o[0] = (_Float16)pow2sgn(sh.x, sg.x);
    o[1] = (_Float16)pow2sgn(sh.y, sg.y);
    o[2] = (_Float16)pow2sgn(sh.z, sg.z);
    o[3] = (_Float16)pow2sgn(sh.w, sg.w);
    *(v4h*)(vh + i) = o;
}

// ---------------- main GEMM: out[N,OUT] = Xh[N,K] * Vh[OUT,K]^T + bias ----------------
__global__ __launch_bounds__(256)
void linshift_gemm(const _Float16* __restrict__ Xh,
                   const _Float16* __restrict__ Vh,
                   const float*    __restrict__ bias,
                   float* __restrict__ out,
                   int N, int K, int OUT)
{
    __shared__ _Float16 As[2][BM * LDT];
    __shared__ _Float16 Bs[2][BN * LDT];

    const int t    = threadIdx.x;
    const int lane = t & 31;
    const int wave = t >> 5;         // 0..7
    const int wm   = wave >> 1;      // 0..3 : wave tile row (32 rows each)
    const int wn   = wave & 1;       // 0..1 : wave tile col (64 cols each)
    const int m0   = blockIdx.y * BM;
    const int n0   = blockIdx.x * BN;

    // ISA 16-bit A/B fragment layout: lane = row(0..15); lane<16 holds K 0-7,16-23;
    // lane>=16 holds K 8-15,24-31.
    const int frow = lane & 15;
    const int fk   = (lane >> 4) * 8;

    // stage one 128x32 f16 tile of A and of B into LDS buffer `buf`
    // (4 b128 transfers per thread per call -> ASYNCcnt += 4 on the async path)
    auto stage = [&](int buf, int k0) {
#pragma unroll
        for (int p = 0; p < 2; ++p) {
            int idx = t + p * 256;        // 0..511
            int row = idx >> 2;           // 0..127
            int seg = (idx & 3) * 8;      // 0,8,16,24
            const _Float16* ga = &Xh[(size_t)(m0 + row) * K + k0 + seg];
            const _Float16* gb = &Vh[(size_t)(n0 + row) * K + k0 + seg];
            _Float16* la = &As[buf][row * LDT + seg];
            _Float16* lb = &Bs[buf][row * LDT + seg];
#ifdef USE_ASYNC_LDS
            __builtin_amdgcn_global_load_async_to_lds_b128(
                (AS1 v4i*)ga, (AS3 v4i*)la, 0, 0);
            __builtin_amdgcn_global_load_async_to_lds_b128(
                (AS1 v4i*)gb, (AS3 v4i*)lb, 0, 0);
#else
            *(v8h*)la = *(const v8h*)ga;
            *(v8h*)lb = *(const v8h*)gb;
#endif
        }
    };

    v8f acc[2][4];
#pragma unroll
    for (int mi = 0; mi < 2; ++mi)
#pragma unroll
        for (int ni = 0; ni < 4; ++ni)
            acc[mi][ni] = v8f{};

    union Frag { v16h v; v8h h[2]; };

    stage(0, 0);  // prologue: fill buffer 0

    for (int k0 = 0; k0 < K; k0 += BK) {
        const int cur = (k0 >> 5) & 1;

        // prefetch next K-step into the other buffer, overlapping this step's math
        if (k0 + BK < K) {
            stage(cur ^ 1, k0 + BK);
#ifdef USE_ASYNC_LDS
            ASYNC_WAIT(4);   // async loads retire in order: <=4 outstanding ==> `cur` landed
#endif
        } else {
#ifdef USE_ASYNC_LDS
            ASYNC_WAIT(0);
#endif
        }
        __syncthreads();

        Frag a[2], b[4];
#pragma unroll
        for (int mi = 0; mi < 2; ++mi) {
            const _Float16* p = &As[cur][(wm * 32 + mi * 16 + frow) * LDT + fk];
            a[mi].h[0] = *(const v8h*)p;          // K = fk .. fk+7
            a[mi].h[1] = *(const v8h*)(p + 16);   // K = fk+16 .. fk+23
        }
#pragma unroll
        for (int ni = 0; ni < 4; ++ni) {
            const _Float16* p = &Bs[cur][(wn * 64 + ni * 16 + frow) * LDT + fk];
            b[ni].h[0] = *(const v8h*)p;
            b[ni].h[1] = *(const v8h*)(p + 16);
        }

#pragma unroll
        for (int mi = 0; mi < 2; ++mi)
#pragma unroll
            for (int ni = 0; ni < 4; ++ni)
                acc[mi][ni] = __builtin_amdgcn_wmma_f32_16x16x32_f16(
                    false, a[mi].v, false, b[ni].v,
                    (short)0, acc[mi][ni], false, false);

        __syncthreads();   // all waves done reading `cur` before it is restaged
    }

    // epilogue: C/D layout -> VGPR i: M = i + 8*(lane>=16), N = lane&15
    const int rhi = (lane >> 4) * 8;
#pragma unroll
    for (int ni = 0; ni < 4; ++ni) {
        int col = n0 + wn * 64 + ni * 16 + frow;
        float bv = bias[col];
#pragma unroll
        for (int mi = 0; mi < 2; ++mi) {
            int r0 = m0 + wm * 32 + mi * 16 + rhi;
#pragma unroll
            for (int i = 0; i < 8; ++i)
                out[(size_t)(r0 + i) * OUT + col] = acc[mi][ni][i] + bv;
        }
    }
}

extern "C" void kernel_launch(void* const* d_in, const int* in_sizes, int n_in,
                              void* d_out, int out_size, void* d_ws, size_t ws_size,
                              hipStream_t stream)
{
    const float* x     = (const float*)d_in[0];
    const float* shift = (const float*)d_in[1];
    const float* sgn   = (const float*)d_in[2];
    const float* bias  = (const float*)d_in[3];
    float* out = (float*)d_out;

    const long OUT = (long)in_sizes[3];            // 4096
    const long K   = (long)in_sizes[1] / OUT;      // 4096
    const long N   = (long)in_sizes[0] / K;        // 8192

    _Float16* xh = (_Float16*)d_ws;
    _Float16* vh = (_Float16*)((char*)d_ws + (size_t)N * K * sizeof(_Float16));

    // pre-pass 1: quantize x to f16
    {
        long nx = N * K;
        long blocks = (nx / 8 + 255) / 256;
        cvt_x_f16<<<dim3((unsigned)blocks), dim3(256), 0, stream>>>(x, xh, nx);
    }
    // pre-pass 2: materialize v = +/-2^k in f16 (exact: k in [-14,0] is normal f16)
    {
        long nv = OUT * K;
        long blocks = (nv / 4 + 255) / 256;
        make_v_f16<<<dim3((unsigned)blocks), dim3(256), 0, stream>>>(shift, sgn, vh, nv);
    }
    // main WMMA GEMM
    {
        dim3 grid((unsigned)(OUT / BN), (unsigned)(N / BM));
        linshift_gemm<<<grid, dim3(256), 0, stream>>>(xh, vh, bias, out,
                                                      (int)N, (int)K, (int)OUT);
    }
}
